// EdgeGNNClassifier_45475113730276
// MI455X (gfx1250) — matrix operand order repaired and checked by reading, not compiled, not run on hardware
//
#include <hip/hip_runtime.h>

// MI455X / gfx1250, wave32. All matrix math via v_wmma_f32_16x16x32_f16.
// B operands pre-swizzled into fragment order (32B contiguous per lane);
// A operands assembled from float4 runs (the ISA A K-map is two 8-float runs).

typedef _Float16 half16 __attribute__((ext_vector_type(16)));
typedef float    float8 __attribute__((ext_vector_type(8)));

#define NCOL 128   // H * Ch
#define CH   32

// ---- order-preserving float <-> uint for atomic segment-max ----
__device__ __forceinline__ unsigned flipf(float x) {
    unsigned u = __float_as_uint(x);
    return (u & 0x80000000u) ? ~u : (u | 0x80000000u);
}
__device__ __forceinline__ float unflipf(unsigned u) {
    return (u & 0x80000000u) ? __uint_as_float(u & 0x7fffffffu) : __uint_as_float(~u);
}

// pack 8 consecutive K-values (two float4s) into fragment elements [base..base+7]
__device__ __forceinline__ void pack8(half16& a, int base, const float4& r0, const float4& r1) {
    a[base + 0] = (_Float16)r0.x; a[base + 1] = (_Float16)r0.y;
    a[base + 2] = (_Float16)r0.z; a[base + 3] = (_Float16)r0.w;
    a[base + 4] = (_Float16)r1.x; a[base + 5] = (_Float16)r1.y;
    a[base + 6] = (_Float16)r1.z; a[base + 7] = (_Float16)r1.w;
}

// =================== weight pre-swizzle: f32 [K,ncols] -> f16 B fragments ===================
// out[((kt*ntc + t)*32 + lane)*16 + j] = W[(kt*32 + 2*(j&8) + 8*(lane>>4) + (j&7)) * ncols + t*16 + (lane&15)]
// (zero-padded for k >= krows). A wave then loads a whole fragment as half16 per lane.
__global__ void k_make_bfrag(const float* __restrict__ W, _Float16* __restrict__ out,
                             int krows, int ncols, int total) {
    int i = blockIdx.x * blockDim.x + threadIdx.x;
    if (i >= total) return;
    int j    = i & 15;
    int lane = (i >> 4) & 31;
    int rest = i >> 9;
    int ntc  = ncols >> 4;
    int t    = rest % ntc;
    int kt   = rest / ntc;
    int k    = kt * 32 + ((j & 8) << 1) + ((lane >> 4) << 3) + (j & 7);
    int c    = t * 16 + (lane & 15);
    out[i] = (k < krows) ? (_Float16)W[k * ncols + c] : (_Float16)0.f;
}

__global__ void k_fill_u32(unsigned* __restrict__ p, unsigned v, int n) {
    int i = blockIdx.x * blockDim.x + threadIdx.x;
    if (i < n) p[i] = v;
}

// =================== node linear: q,k,v,skip via WMMA ===================
// X:[N,32] f32. One wave per 16-row tile; A fragment loaded once (4x float4),
// reused across 8+8+8 column tiles of Wq/Wk/Wv and 2 tiles of Ws.
__global__ void k_node_linear(const float* __restrict__ X,
                              const _Float16* __restrict__ WqF,
                              const _Float16* __restrict__ WkF,
                              const _Float16* __restrict__ WvF,
                              const _Float16* __restrict__ WsF,
                              const float* __restrict__ bq,
                              const float* __restrict__ bk,
                              const float* __restrict__ bv,
                              const float* __restrict__ bs,
                              float* __restrict__ q, float* __restrict__ k,
                              float* __restrict__ v, float* __restrict__ skip,
                              int ntiles)
{
    int gid  = blockIdx.x * blockDim.x + threadIdx.x;
    int wave = gid >> 5;
    int lane = gid & 31;
    if (wave >= ntiles) return;               // wave-uniform: EXEC all-ones inside
    int khalf   = lane >> 4;
    int coln    = lane & 15;
    int rowbase = wave << 4;

    const float* Xr = X + (size_t)(rowbase + coln) * 32;
    half16 a;
    {   // K runs [8kh, 8kh+7] and [16+8kh, 16+8kh+7]
        float4 r0 = *(const float4*)(Xr + 8 * khalf);
        float4 r1 = *(const float4*)(Xr + 8 * khalf + 4);
        float4 r2 = *(const float4*)(Xr + 16 + 8 * khalf);
        float4 r3 = *(const float4*)(Xr + 16 + 8 * khalf + 4);
        pack8(a, 0, r0, r1);
        pack8(a, 8, r2, r3);
    }

    const half16* Fq = (const half16*)WqF;
    const half16* Fk = (const half16*)WkF;
    const half16* Fv = (const half16*)WvF;
    const half16* Fs = (const half16*)WsF;

    const half16* Fmat[3] = {Fq, Fk, Fv};
    const float*  Bvec[3] = {bq, bk, bv};
    float*        Omat[3] = {q, k, v};
#pragma unroll
    for (int m = 0; m < 3; ++m) {
        const half16* F = Fmat[m];
        const float*  B = Bvec[m];
        float*        O = Omat[m];
#pragma unroll
        for (int t = 0; t < 8; ++t) {
            half16 b = F[t * 32 + lane];           // contiguous 32B fragment load
            float8 c = {};
            c = __builtin_amdgcn_wmma_f32_16x16x32_f16(false, a, false, b, (short)0, c, false, false);
            float bb = B[t * 16 + coln];
            float* Ob = O + (size_t)(rowbase + (khalf << 3)) * NCOL + t * 16 + coln;
#pragma unroll
            for (int i = 0; i < 8; ++i)
                Ob[(size_t)i * NCOL] = c[i] + bb;
        }
    }
    // skip = X @ Ws + bs   ([32,32] -> 2 column tiles)
#pragma unroll
    for (int t = 0; t < 2; ++t) {
        half16 b = Fs[t * 32 + lane];
        float8 c = {};
        c = __builtin_amdgcn_wmma_f32_16x16x32_f16(false, a, false, b, (short)0, c, false, false);
        float bb = bs[t * 16 + coln];
        float* Ob = skip + (size_t)(rowbase + (khalf << 3)) * CH + t * 16 + coln;
#pragma unroll
        for (int i = 0; i < 8; ++i)
            Ob[(size_t)i * CH] = c[i] + bb;
    }
}

// =================== edge feature GEMM: e = ea @ We (K padded 16->32) ===================
__global__ void k_edge_e(const float* __restrict__ ea, const _Float16* __restrict__ WeF,
                         _Float16* __restrict__ eh, int ntiles)
{
    int gid  = blockIdx.x * blockDim.x + threadIdx.x;
    int wave = gid >> 5;
    int lane = gid & 31;
    if (wave >= ntiles) return;
    int khalf   = lane >> 4;
    int coln    = lane & 15;
    int rowbase = wave << 4;

    const float* Er = ea + (size_t)(rowbase + coln) * 16;
    half16 a;
    {   // only K run [8kh, 8kh+7] is real; K>=16 is zero padding
        float4 r0 = *(const float4*)(Er + 8 * khalf);
        float4 r1 = *(const float4*)(Er + 8 * khalf + 4);
        pack8(a, 0, r0, r1);
#pragma unroll
        for (int j = 8; j < 16; ++j) a[j] = (_Float16)0.f;
    }
    const half16* F = (const half16*)WeF;
#pragma unroll
    for (int t = 0; t < 8; ++t) {
        half16 b = F[t * 32 + lane];
        float8 c = {};
        c = __builtin_amdgcn_wmma_f32_16x16x32_f16(false, a, false, b, (short)0, c, false, false);
        _Float16* Ob = eh + (size_t)(rowbase + (khalf << 3)) * NCOL + t * 16 + coln;
#pragma unroll
        for (int i = 0; i < 8; ++i)
            Ob[(size_t)i * NCOL] = (_Float16)c[i];
    }
}

// =================== wave-per-edge: alpha + segment max ===================
__global__ void k_edge_alpha(const float* __restrict__ q, const float* __restrict__ k,
                             const _Float16* __restrict__ eh,
                             const int* __restrict__ src, const int* __restrict__ dst,
                             float* __restrict__ alpha, unsigned* __restrict__ mflip,
                             int E, float scale)
{
    int gid  = blockIdx.x * blockDim.x + threadIdx.x;
    int e    = gid >> 5;
    int lane = gid & 31;
    if (e >= E) return;
    int sn = src[e], dn = dst[e];
#pragma unroll
    for (int h = 0; h < 4; ++h) {
        int c = h * 32 + lane;                       // coalesced 128B per head
        float kj = k[(size_t)sn * NCOL + c] + (float)eh[(size_t)e * NCOL + c];
        float p  = q[(size_t)dn * NCOL + c] * kj;
#pragma unroll
        for (int m = 16; m >= 1; m >>= 1) p += __shfl_xor(p, m, 32);
        p *= scale;
        if (lane == h) {
            alpha[(size_t)e * 4 + h] = p;
            atomicMax(&mflip[dn * 4 + h], flipf(p));
        }
    }
}

// =================== wave-per-edge: exp, sum, weighted V aggregation ===================
__global__ void k_edge_aggregate(const float* __restrict__ v, const _Float16* __restrict__ eh,
                                 const float* __restrict__ alpha, const unsigned* __restrict__ mflip,
                                 const int* __restrict__ src, const int* __restrict__ dst,
                                 float* __restrict__ ssum, float* __restrict__ acc, int E)
{
    int gid  = blockIdx.x * blockDim.x + threadIdx.x;
    int e    = gid >> 5;
    int lane = gid & 31;
    if (e >= E) return;
    int sn = src[e], dn = dst[e];
    float aexp[4];
#pragma unroll
    for (int h = 0; h < 4; ++h) {
        float al = alpha[(size_t)e * 4 + h];
        float mm = unflipf(mflip[dn * 4 + h]);
        aexp[h] = __expf(al - mm);
    }
    if (lane < 4) atomicAdd(&ssum[dn * 4 + lane], aexp[lane]);
#pragma unroll
    for (int h = 0; h < 4; ++h) {
        int c = h * 32 + lane;
        float vj = v[(size_t)sn * NCOL + c] + (float)eh[(size_t)e * NCOL + c];
        atomicAdd(&acc[(size_t)dn * NCOL + c], aexp[h] * vj);   // 25.6MB buffer: L2 resident
    }
}

// =================== node finalize: normalize, head-mean, skip, ReLU ===================
__global__ void k_node_finalize(const float* __restrict__ acc, const float* __restrict__ ssum,
                                const float* __restrict__ skip, float* __restrict__ out, int N)
{
    int t = blockIdx.x * blockDim.x + threadIdx.x;
    if (t >= N * CH) return;
    int n = t >> 5, c = t & 31;
    float o = 0.f;
#pragma unroll
    for (int h = 0; h < 4; ++h)
        o += acc[(size_t)n * NCOL + h * 32 + c] / (ssum[n * 4 + h] + 1e-16f);
    o = o * 0.25f + skip[t];
    out[t] = fmaxf(o, 0.f);
}

// =================== edge MLP: [h[src], ea, h[dst]] (80->96 pad) @ Wm1, ReLU, @ Wm2 ===================
// Each aligned 8-run of the 96-wide ef vector lies wholly inside one segment.
__device__ __forceinline__ void load_run8(half16& a, int base, int kbase,
                                          const float* hs, const float* eaR, const float* hd) {
    if (kbase >= 80) {
#pragma unroll
        for (int j = 0; j < 8; ++j) a[base + j] = (_Float16)0.f;
        return;
    }
    const float* p; int o;
    if (kbase < 32)      { p = hs;  o = kbase; }
    else if (kbase < 48) { p = eaR; o = kbase - 32; }
    else                 { p = hd;  o = kbase - 48; }
    float4 r0 = *(const float4*)(p + o);
    float4 r1 = *(const float4*)(p + o + 4);
    pack8(a, base, r0, r1);
}

__global__ void k_mlp(const float* __restrict__ h, const float* __restrict__ ea,
                      const int* __restrict__ src, const int* __restrict__ dst,
                      const _Float16* __restrict__ Wm1F, const float* __restrict__ bm1,
                      const float* __restrict__ Wm2, const float* __restrict__ bm2,
                      float* __restrict__ out, int ntiles)
{
    int gid  = blockIdx.x * blockDim.x + threadIdx.x;
    int wave = gid >> 5;
    int lane = gid & 31;
    if (wave >= ntiles) return;
    int khalf = lane >> 4;
    int coln  = lane & 15;
    int e     = wave * 16 + coln;                    // this lane's edge row of the A tile
    int sn = src[e], dn = dst[e];
    const float* hs  = h + (size_t)sn * CH;
    const float* hd  = h + (size_t)dn * CH;
    const float* eaR = ea + (size_t)e * 16;
    const half16* F  = (const half16*)Wm1F;          // 3 ktiles x 2 coltiles

    float8 c0 = {}, c1 = {};
#pragma unroll
    for (int kt = 0; kt < 3; ++kt) {                 // K = 96 in 3 steps of 32
        half16 a;
        load_run8(a, 0, kt * 32 + 8 * khalf, hs, eaR, hd);
        load_run8(a, 8, kt * 32 + 16 + 8 * khalf, hs, eaR, hd);
        half16 b0 = F[(kt * 2 + 0) * 32 + lane];
        half16 b1 = F[(kt * 2 + 1) * 32 + lane];
        c0 = __builtin_amdgcn_wmma_f32_16x16x32_f16(false, a, false, b0, (short)0, c0, false, false);
        c1 = __builtin_amdgcn_wmma_f32_16x16x32_f16(false, a, false, b1, (short)0, c1, false, false);
    }
    // second layer: out[m] = sum_c relu(hid[m][c]+bm1[c]) * Wm2[c] + bm2
    float b1a = bm1[coln], b1b = bm1[16 + coln];
    float w2a = Wm2[coln], w2b = Wm2[16 + coln];
    float r[8];
#pragma unroll
    for (int i = 0; i < 8; ++i) {
        float s = fmaxf(c0[i] + b1a, 0.f) * w2a + fmaxf(c1[i] + b1b, 0.f) * w2b;
#pragma unroll
        for (int m = 1; m <= 8; m <<= 1) s += __shfl_xor(s, m, 32);  // reduce 16 lanes of half
        r[i] = s;
    }
    if (coln == 0) {
        float bb = bm2[0];
#pragma unroll
        for (int i = 0; i < 8; ++i)
            out[wave * 16 + i + (khalf << 3)] = r[i] + bb;
    }
}

// =================== host launcher ===================
extern "C" void kernel_launch(void* const* d_in, const int* in_sizes, int n_in,
                              void* d_out, int out_size, void* d_ws, size_t ws_size,
                              hipStream_t stream) {
    (void)in_sizes; (void)n_in; (void)out_size; (void)ws_size;
    const int N = 50000, E = 800000;

    const float* x   = (const float*)d_in[0];
    const float* ea  = (const float*)d_in[1];
    const int*  eidx = (const int*)d_in[2];
    const int* srcI = eidx;
    const int* dstI = eidx + E;
    const float* Wq[2] = {(const float*)d_in[3],  (const float*)d_in[12]};
    const float* bq[2] = {(const float*)d_in[4],  (const float*)d_in[13]};
    const float* Wk[2] = {(const float*)d_in[5],  (const float*)d_in[14]};
    const float* bk[2] = {(const float*)d_in[6],  (const float*)d_in[15]};
    const float* Wv[2] = {(const float*)d_in[7],  (const float*)d_in[16]};
    const float* bv[2] = {(const float*)d_in[8],  (const float*)d_in[17]};
    const float* We[2] = {(const float*)d_in[9],  (const float*)d_in[18]};
    const float* Wsk[2]= {(const float*)d_in[10], (const float*)d_in[19]};
    const float* bsk[2]= {(const float*)d_in[11], (const float*)d_in[20]};
    const float* Wm1 = (const float*)d_in[21];
    const float* bm1 = (const float*)d_in[22];
    const float* Wm2 = (const float*)d_in[23];
    const float* bm2 = (const float*)d_in[24];

    char* ws = (char*)d_ws;
    size_t off = 0;
    auto take = [&](size_t bytes) -> char* {
        off = (off + 255) & ~(size_t)255;
        char* p = ws + off; off += bytes; return p;
    };
    // fragment-ordered f16 weights (K padded to multiples of 32)
    _Float16 *WqF[2], *WkF[2], *WvF[2], *WsF[2], *WeF[2];
    for (int L = 0; L < 2; ++L) {
        WqF[L] = (_Float16*)take(1 * 8 * 512 * 2);   // 1 ktile x 8 coltiles
        WkF[L] = (_Float16*)take(1 * 8 * 512 * 2);
        WvF[L] = (_Float16*)take(1 * 8 * 512 * 2);
        WsF[L] = (_Float16*)take(1 * 2 * 512 * 2);   // 1 ktile x 2 coltiles
        WeF[L] = (_Float16*)take(1 * 8 * 512 * 2);   // K 16 padded to 32
    }
    _Float16* Wm1F = (_Float16*)take(3 * 2 * 512 * 2);  // K 80 padded to 96
    float* qb    = (float*)take((size_t)N * NCOL * 4);
    float* kb    = (float*)take((size_t)N * NCOL * 4);
    float* vb    = (float*)take((size_t)N * NCOL * 4);
    float* skipb = (float*)take((size_t)N * CH * 4);
    _Float16* ehb = (_Float16*)take((size_t)E * NCOL * 2);
    float* alphab = (float*)take((size_t)E * 4 * 4);
    unsigned* mb  = (unsigned*)take((size_t)N * 4 * 4);
    float* sb     = (float*)take((size_t)N * 4 * 4);
    float* accb   = (float*)take((size_t)N * NCOL * 4);
    float* h1     = (float*)take((size_t)N * CH * 4);
    float* h2     = (float*)take((size_t)N * CH * 4);

    auto mk = [&](const float* W, _Float16* F, int krows, int ncols, int nktiles) {
        int total = nktiles * (ncols / 16) * 512;
        k_make_bfrag<<<(total + 255) / 256, 256, 0, stream>>>(W, F, krows, ncols, total);
    };
    for (int L = 0; L < 2; ++L) {
        mk(Wq[L], WqF[L], 32, 128, 1);
        mk(Wk[L], WkF[L], 32, 128, 1);
        mk(Wv[L], WvF[L], 32, 128, 1);
        mk(Wsk[L], WsF[L], 32, 32, 1);
        mk(We[L], WeF[L], 16, 128, 1);
    }
    mk(Wm1, Wm1F, 80, 32, 3);

    const float scale = 0.17677669529663687f;  // 1/sqrt(Dh=32)
    const int ntN = N / 16, ntE = E / 16;
    const float* layer_in = x;
    float* layer_out[2] = {h1, h2};

    for (int L = 0; L < 2; ++L) {
        k_node_linear<<<(ntN * 32 + 255) / 256, 256, 0, stream>>>(
            layer_in, WqF[L], WkF[L], WvF[L], WsF[L],
            bq[L], bk[L], bv[L], bsk[L], qb, kb, vb, skipb, ntN);
        k_edge_e<<<(ntE * 32 + 255) / 256, 256, 0, stream>>>(ea, WeF[L], ehb, ntE);
        k_fill_u32<<<(N * 4 + 255) / 256, 256, 0, stream>>>(mb, 0x007FFFFFu, N * 4);   // flip(-inf)
        k_fill_u32<<<(N * 4 + 255) / 256, 256, 0, stream>>>((unsigned*)sb, 0u, N * 4);
        k_fill_u32<<<(N * NCOL + 255) / 256, 256, 0, stream>>>((unsigned*)accb, 0u, N * NCOL);
        k_edge_alpha<<<((size_t)E * 32 + 255) / 256, 256, 0, stream>>>(
            qb, kb, ehb, srcI, dstI, alphab, mb, E, scale);
        k_edge_aggregate<<<((size_t)E * 32 + 255) / 256, 256, 0, stream>>>(
            vb, ehb, alphab, mb, srcI, dstI, sb, accb, E);
        k_node_finalize<<<(N * CH + 255) / 256, 256, 0, stream>>>(accb, sb, skipb, layer_out[L], N);
        layer_in = layer_out[L];
    }

    k_mlp<<<(ntE * 32 + 255) / 256, 256, 0, stream>>>(
        h2, ea, srcI, dstI, Wm1F, bm1, Wm2, bm2, (float*)d_out, ntE);
}
